// Model_39676907887621
// MI455X (gfx1250) — matrix operand order repaired and checked by reading, not compiled, not run on hardware
//
#include <hip/hip_runtime.h>
#include <hip/hip_bf16.h>
#include <stdint.h>

typedef __attribute__((ext_vector_type(16))) __bf16 v16bf;
typedef __attribute__((ext_vector_type(8)))  __bf16 v8bf;
typedef __attribute__((ext_vector_type(4)))  __bf16 v4bf;
typedef __attribute__((ext_vector_type(8)))  float  v8f;
typedef __attribute__((ext_vector_type(4)))  float  v4f;

#define S_LEN   2048
#define D_DIM   64
#define QBLK    128     // queries per block (8 waves x 16)
#define KTILE   128     // keys staged per LDS tile
#define SCALE_F 10000.0f
#define KEEP_F  0.8f
// scores pre-scaled into the exp2 domain: s2 = s * SCALE * log2(e)
#define C2LOG2E (SCALE_F * 1.44269504088896340736f)

__global__ __launch_bounds__(256)
void fattn_bf16_wmma(const float* __restrict__ x1,
                     const float* __restrict__ x2,
                     float* __restrict__ out) {
    // LDS: K tile row-major [key][d] (A-fragments of S^T = K x Q^T),
    //      V tile transposed [d][key] (B-fragments of O = P x V).
    __shared__ __align__(32) __bf16 sK [KTILE * D_DIM];
    __shared__ __align__(32) __bf16 sVt[D_DIM * KTILE];

    const int qb   = blockIdx.x;          // 0..15  query block
    const int bh   = blockIdx.y;          // 0..31  (b*H + h)
    const int tid  = threadIdx.x;
    const int lane = tid & 31;
    const int wave = tid >> 5;
    const int n    = lane & 15;           // column index within fragment
    const int hi   = lane >> 4;           // half-wave select

    const size_t bh_off = (size_t)bh * S_LEN * D_DIM;
    const float* Qg = x1 + bh_off;
    const float* Kg = x2 + bh_off;        // K == V == x2
    float*       Og = out + bh_off;

    const int qrow0 = qb * QBLK + wave * 16;

    // ---- Q^T B-fragments (held in registers for the whole kernel) ----
    // Fragment F covers d = F*32..F*32+31. B layout: lanes 0-15 hold rows
    // d = F*32+0..15 (col = query n), lanes 16-31 hold d = F*32+16..31;
    // each lane gets 16 *contiguous* d values -> 4x b128 global loads.
    v16bf Qf[2];
    {
        const float* qrow = Qg + (size_t)(qrow0 + n) * D_DIM;
        #pragma unroll
        for (int F = 0; F < 2; ++F) {
            const v4f* p = (const v4f*)(qrow + F * 32 + hi * 16);
            v4f a = p[0], b = p[1], c = p[2], d = p[3];
            v16bf f;
            #pragma unroll
            for (int e = 0; e < 4; ++e) {
                f[e]      = (__bf16)a[e];
                f[4 + e]  = (__bf16)b[e];
                f[8 + e]  = (__bf16)c[e];
                f[12 + e] = (__bf16)d[e];
            }
            Qf[F] = f;
        }
    }

    // ---- online softmax state (exp2 domain): lane <-> query column ----
    float m = -3.0e38f;
    float l = 0.0f;
    v8f O[4];
    #pragma unroll
    for (int td = 0; td < 4; ++td) O[td] = (v8f)(0.0f);

    const uint32_t keepThr = 0xCCCCCCCCu;   // keep = 0.8 of 2^32

    for (int kb = 0; kb < S_LEN; kb += KTILE) {
        // ---- cooperative stage: f32 global -> bf16 LDS (K row-major + V^T) ----
        __syncthreads();
        {
            const v4f* src = (const v4f*)(Kg + (size_t)kb * D_DIM);
            #pragma unroll
            for (int i = 0; i < (KTILE * D_DIM) / (256 * 4); ++i) {
                const int idx = i * 256 + tid;       // coalesced b128 loads
                const v4f v = src[idx];
                const int kk = idx >> 4;             // key row in tile
                const int dd = (idx & 15) * 4;       // d base
                const __bf16 b0 = (__bf16)v[0], b1 = (__bf16)v[1];
                const __bf16 b2 = (__bf16)v[2], b3 = (__bf16)v[3];
                *(v4bf*)&sK[kk * D_DIM + dd] = (v4bf){b0, b1, b2, b3};
                sVt[(dd + 0) * KTILE + kk] = b0;
                sVt[(dd + 1) * KTILE + kk] = b1;
                sVt[(dd + 2) * KTILE + kk] = b2;
                sVt[(dd + 3) * KTILE + kk] = b3;
            }
        }
        __syncthreads();

        // speculative prefetch of next tile (global_prefetch_b8)
        {
            const int kn = (kb + KTILE < S_LEN) ? (kb + KTILE) : kb;
            __builtin_prefetch(Kg + (size_t)kn * D_DIM + ((size_t)tid << 2), 0, 1);
        }

        // ---- 4 chunks of 32 keys ----
        #pragma unroll 1
        for (int c = 0; c < 4; ++c) {
            // S^T for the two 16-key sub-tiles: A = K rows, B = Q^T.
            // C layout: lane = query col, regs = keys (lo half keys r, hi half r+8).
            v8f s01[2];
            #pragma unroll
            for (int t = 0; t < 2; ++t) {
                const __bf16* kp = &sK[(c * 32 + t * 16 + n) * D_DIM];
                v8f acc = (v8f)(0.0f);
                #pragma unroll
                for (int F = 0; F < 2; ++F) {
                    const int d0 = F * 32 + hi * 16;
                    const v8bf klo = *(const v8bf*)(kp + d0);
                    const v8bf khi = *(const v8bf*)(kp + d0 + 8);
                    v16bf af;
                    #pragma unroll
                    for (int e = 0; e < 8; ++e) { af[e] = klo[e]; af[8 + e] = khi[e]; }
                    acc = __builtin_amdgcn_wmma_f32_16x16x32_bf16(
                        false, af, false, Qf[F], (short)0, acc, false, false);
                }
                s01[t] = acc;
            }

            // scores scaled straight into the exp2 domain
            float sv[16];
            #pragma unroll
            for (int r = 0; r < 8; ++r) {
                sv[r]     = s01[0][r] * C2LOG2E;
                sv[8 + r] = s01[1][r] * C2LOG2E;
            }
            float lm = sv[0];
            #pragma unroll
            for (int r = 1; r < 16; ++r) lm = fmaxf(lm, sv[r]);
            lm = fmaxf(lm, __shfl_xor(lm, 16, 32));
            const float mnew  = fmaxf(m, lm);
            const float alpha = __builtin_amdgcn_exp2f(m - mnew);

            // exp2, denominator accumulation (pre-dropout), then dropout as a
            // pure select (1/keep is folded into the epilogue scale).
            // hash: Knuth multiplicative + one xor-shift, threshold compare.
            const uint32_t idx0 =
                (((uint32_t)(bh * S_LEN + qrow0 + n)) << 11) +
                (uint32_t)(kb + c * 32 + hi * 8);
            float psum = 0.0f;
            v16bf P;
            #pragma unroll
            for (int r = 0; r < 8; ++r) {
                const float e0 = __builtin_amdgcn_exp2f(sv[r]     - mnew);
                const float e1 = __builtin_amdgcn_exp2f(sv[8 + r] - mnew);
                psum += e0 + e1;
                uint32_t h0 = (idx0 + (uint32_t)r) * 2654435761u;
                uint32_t h1 = (idx0 + (uint32_t)(r + 16)) * 2654435761u;
                h0 ^= h0 >> 17;
                h1 ^= h1 >> 17;
                P[r]     = (__bf16)((h0 < keepThr) ? e0 : 0.0f);
                P[8 + r] = (__bf16)((h1 < keepThr) ? e1 : 0.0f);
            }
            psum += __shfl_xor(psum, 16, 32);
            l = l * alpha + psum;
            m = mnew;

            // rescale O: row (query) M = r + hi*8 lives in lane r + hi*8
            #pragma unroll
            for (int r = 0; r < 8; ++r) {
                const float ar = __shfl(alpha, r + hi * 8, 32);
                #pragma unroll
                for (int td = 0; td < 4; ++td) O[td][r] *= ar;
            }

            // O += P x V over four 16-wide d tiles; V B-frag = two 16B reads
            // (keys hi*8..+7 and 16+hi*8..+7) of transposed V in LDS.
            #pragma unroll
            for (int td = 0; td < 4; ++td) {
                const __bf16* vp = &sVt[(td * 16 + n) * KTILE + c * 32 + hi * 8];
                const v8bf vlo = *(const v8bf*)vp;
                const v8bf vhi = *(const v8bf*)(vp + 16);
                v16bf vf;
                #pragma unroll
                for (int e = 0; e < 8; ++e) { vf[e] = vlo[e]; vf[8 + e] = vhi[e]; }
                O[td] = __builtin_amdgcn_wmma_f32_16x16x32_bf16(
                    false, P, false, vf, (short)0, O[td], false, false);
            }
        }
    }

    // ---- epilogue: (1/l)*(1/keep) per query row, store fp32 output ----
    const float scaleOut = (1.0f / l) * (1.0f / KEEP_F);
    #pragma unroll
    for (int r = 0; r < 8; ++r) {
        const float lr = __shfl(scaleOut, r + hi * 8, 32);
        float* orow = Og + (size_t)(qrow0 + r + hi * 8) * D_DIM;
        #pragma unroll
        for (int td = 0; td < 4; ++td) orow[td * 16 + n] = O[td][r] * lr;
    }
}

extern "C" void kernel_launch(void* const* d_in, const int* in_sizes, int n_in,
                              void* d_out, int out_size, void* d_ws, size_t ws_size,
                              hipStream_t stream) {
    (void)in_sizes; (void)n_in; (void)out_size; (void)d_ws; (void)ws_size;
    const float* x1 = (const float*)d_in[0];
    const float* x2 = (const float*)d_in[1];
    float* out = (float*)d_out;
    dim3 grid(S_LEN / QBLK, 2 * 16);   // (16 query blocks, B*H = 32)
    dim3 block(256);                   // 8 wave32 waves
    fattn_bf16_wmma<<<grid, block, 0, stream>>>(x1, x2, out);
}
